// SelfAttention_88785563943378
// MI455X (gfx1250) — matrix operand order repaired and checked by reading, not compiled, not run on hardware
//
#include <hip/hip_runtime.h>

// ---------------------------------------------------------------------------
// Self-attention forward for MI455X (gfx1250, wave32, WMMA).
// Pipeline: cvt(x,wT) -> QKV gemm -> scores gemm -> softmax -> PV gemm -> out
// GEMM core: NT (A[M][K] x Bt[N][K]) with v_wmma_f32_16x16x32_bf16,
// double-buffered LDS fed by global_load_async_to_lds_b128 (ASYNCcnt).
// ---------------------------------------------------------------------------

typedef __attribute__((ext_vector_type(16))) __bf16 v16bf;
typedef __attribute__((ext_vector_type(8)))  float  v8f;

union FragBF { v16bf v; uint4 u[2]; };

#define BM 128            // block tile M
#define BN 256            // block tile N
#define BK 32             // K step
#define LDSS 40           // 32 + 8 pad (bf16 elems) -> rows stay 16B aligned

// Low 32 bits of a flat shared pointer == LDS byte offset (aperture layout).
__device__ __forceinline__ unsigned lds_off(const void* p)
{
    return (unsigned)(unsigned long long)p;
}

// 16B per lane, global -> LDS, tracked by ASYNCcnt (CDNA5 async copy path).
__device__ __forceinline__ void async_cp16(unsigned dst_lds, const __bf16* g)
{
    asm volatile("global_load_async_to_lds_b128 %0, %1, off"
                 :: "v"(dst_lds), "v"(g) : "memory");
}

// NT GEMM core: C[M][N] = A[M][K] * Bt[N][K]^T
// 8 waves (2M x 4N), wave tile 64x64 => 4x4 wmma accums (16 wmma / K-step).
template <typename Epi>
__device__ __forceinline__ void gemm_nt_core(const __bf16* __restrict__ A,
                                             const __bf16* __restrict__ Bt,
                                             int lda, int ldb, int K, Epi epi)
{
    __shared__ __bf16 smA[2][BM][LDSS];   // 2 x 128 x 40 x 2B = 20 KB
    __shared__ __bf16 smB[2][BN][LDSS];   // 2 x 256 x 40 x 2B = 40 KB

    const int tid  = threadIdx.x;
    const int lane = tid & 31;
    const int wv   = tid >> 5;          // wave 0..7
    const int hl   = lane >> 4;         // half-wave 0/1 (K-split in frags)
    const int lr   = lane & 15;         // row/col within 16

    const int m0 = (wv >> 2) * 64;      // wave M origin within block tile
    const int n0 = (wv & 3) * 64;       // wave N origin

    const long blockM = (long)blockIdx.y * BM;
    const long blockN = (long)blockIdx.x * BN;

    // staging coords: 4 threads per row, 8 bf16 (16B) per thread
    const int gr = tid >> 2;            // 0..63
    const int gc = (tid & 3) * 8;       // 0,8,16,24

    const __bf16* agBase = A  + (blockM + gr) * (long)lda + gc;
    const __bf16* bgBase = Bt + (blockN + gr) * (long)ldb + gc;

    auto issue = [&](int k0, int buf) {  // 6 async 16B copies per thread
        const __bf16* ag = agBase + k0;
        const __bf16* bg = bgBase + k0;
        async_cp16(lds_off(&smA[buf][gr      ][gc]), ag);
        async_cp16(lds_off(&smA[buf][gr +  64][gc]), ag +  64l * lda);
        async_cp16(lds_off(&smB[buf][gr      ][gc]), bg);
        async_cp16(lds_off(&smB[buf][gr +  64][gc]), bg +  64l * ldb);
        async_cp16(lds_off(&smB[buf][gr + 128][gc]), bg + 128l * ldb);
        async_cp16(lds_off(&smB[buf][gr + 192][gc]), bg + 192l * ldb);
    };

    v8f acc[4][4] = {};
    const int nk = K / BK;

    issue(0, 0);

    for (int kt = 0; kt < nk; ++kt) {
        const int buf = kt & 1;
        if (kt + 1 < nk) {
            issue((kt + 1) * BK, buf ^ 1);       // prefetch next tile
            // 12 outstanding; <=6 left means tile kt fully landed (in-order)
            asm volatile("s_wait_asynccnt 0x6" ::: "memory");
        } else {
            asm volatile("s_wait_asynccnt 0x0" ::: "memory");
        }
        __syncthreads();                          // tile kt visible to all

        // A frags: lanes 0-15: K 0-15 ; lanes 16-31: K 16-31
        v16bf afr[4];
#pragma unroll
        for (int m = 0; m < 4; ++m) {
            const __bf16* p = &smA[buf][m0 + m * 16 + lr][hl * 16];
            FragBF f;
            f.u[0] = *(const uint4*)p;
            f.u[1] = *(const uint4*)(p + 8);
            afr[m] = f.v;
        }
        // B frags: lanes 0-15: K 0-15 ; lanes 16-31: K 16-31 (Bt K-contig)
        v16bf bfr[4];
#pragma unroll
        for (int n = 0; n < 4; ++n) {
            const __bf16* p = &smB[buf][n0 + n * 16 + lr][hl * 16];
            FragBF f;
            f.u[0] = *(const uint4*)p;
            f.u[1] = *(const uint4*)(p + 8);
            bfr[n] = f.v;
        }

#pragma unroll
        for (int m = 0; m < 4; ++m)
#pragma unroll
            for (int n = 0; n < 4; ++n)
                acc[m][n] = __builtin_amdgcn_wmma_f32_16x16x32_bf16(
                    false, afr[m], false, bfr[n], (short)0, acc[m][n],
                    false, false);

        __syncthreads();   // all waves done reading buf before it's refilled
    }

    // C layout: VGPR r -> M = r + 8*hl, N = lr
#pragma unroll
    for (int m = 0; m < 4; ++m)
#pragma unroll
        for (int n = 0; n < 4; ++n)
#pragma unroll
            for (int r = 0; r < 8; ++r) {
                long row = blockM + m0 + m * 16 + hl * 8 + r;
                long col = blockN + n0 + n * 16 + lr;
                epi(row, col, acc[m][n][r]);
            }
}

// --------------------------- conversion kernels ----------------------------

__global__ void __launch_bounds__(256)
k_cvt_x_bf16(const float* __restrict__ x, __bf16* __restrict__ xbf)
{
    long i = ((long)blockIdx.x * 256 + threadIdx.x) * 4;
    float4 f = *(const float4*)(x + i);
    union { __bf16 h[4]; uint2 u; } o;
    o.h[0] = (__bf16)f.x; o.h[1] = (__bf16)f.y;
    o.h[2] = (__bf16)f.z; o.h[3] = (__bf16)f.w;
    *(uint2*)(xbf + i) = o.u;
}

// w[R][C] (fp32, row-major) -> wT[C][R] (bf16): makes every GEMM an NT gemm.
__global__ void __launch_bounds__(256)
k_cvt_transpose_bf16(const float* __restrict__ w, __bf16* __restrict__ wT,
                     int R, int C)
{
    long i = (long)blockIdx.x * 256 + threadIdx.x;
    long r = i / C, c = i % C;
    wT[c * (long)R + r] = (__bf16)w[i];
}

// ------------------------------ GEMM kernels -------------------------------

// QKV = x @ w_qkv + b ; Q scaled by 1/sqrt(1024)=1/32 ; V written transposed.
__global__ void __launch_bounds__(256)
k_gemm_qkv(const __bf16* __restrict__ xbf, const __bf16* __restrict__ wqkvT,
           const float* __restrict__ bias, __bf16* __restrict__ qbf,
           __bf16* __restrict__ kbf, __bf16* __restrict__ vtbf)
{
    auto epi = [&](long row, long col, float v) {
        v += bias[col];
        long b = row >> 11;       // row = b*2048 + s
        long s = row & 2047;
        if (col < 1024) {
            qbf[(row << 10) + col] = (__bf16)(v * 0.03125f);
        } else if (col < 2048) {
            kbf[(row << 10) + (col - 1024)] = (__bf16)v;
        } else {
            vtbf[((b << 10) + (col - 2048)) * 2048 + s] = (__bf16)v;  // [b,d,t]
        }
    };
    gemm_nt_core(xbf, wqkvT, 1024, 1024, 1024, epi);
}

// scores[b] = Qs @ K^T (scale folded into Q), fp32 into L2-resident ws
__global__ void __launch_bounds__(256)
k_gemm_scores(const __bf16* __restrict__ qbf, const __bf16* __restrict__ kbf,
              float* __restrict__ sc)
{
    long b = blockIdx.z;
    const __bf16* A  = qbf + b * 2048 * 1024;
    const __bf16* Bt = kbf + b * 2048 * 1024;
    float* S = sc + b * 2048 * 2048;
    auto epi = [&](long row, long col, float v) { S[(row << 11) + col] = v; };
    gemm_nt_core(A, Bt, 1024, 1024, 1024, epi);
}

// O[b] = P @ V  (A = P[s][t], Bt = vt[d][t])
__global__ void __launch_bounds__(256)
k_gemm_pv(const __bf16* __restrict__ pbf, const __bf16* __restrict__ vtbf,
          __bf16* __restrict__ obf)
{
    long b = blockIdx.z;
    const __bf16* A  = pbf  + b * 2048 * 2048;
    const __bf16* Bt = vtbf + b * 1024 * 2048;
    __bf16* O = obf + b * 2048 * 1024;
    auto epi = [&](long row, long col, float v) {
        O[(row << 10) + col] = (__bf16)v;
    };
    gemm_nt_core(A, Bt, 2048, 2048, 2048, epi);
}

// out = O @ w_out + b_out  (fp32 result)
__global__ void __launch_bounds__(256)
k_gemm_out(const __bf16* __restrict__ obf, const __bf16* __restrict__ woutT,
           const float* __restrict__ bias, float* __restrict__ out)
{
    auto epi = [&](long row, long col, float v) {
        out[(row << 10) + col] = v + bias[col];
    };
    gemm_nt_core(obf, woutT, 1024, 1024, 1024, epi);
}

// ------------------------------- softmax -----------------------------------
// One 256-thread block per row of 2048 fp32 scores -> bf16 probabilities.
__global__ void __launch_bounds__(256)
k_softmax(const float* __restrict__ sc, __bf16* __restrict__ pbf)
{
    const long row = blockIdx.x;
    const float* r = sc + (row << 11);
    __bf16* o = pbf + (row << 11);
    const int tid = threadIdx.x;

    float4 x0 = *(const float4*)(r + tid * 8);
    float4 x1 = *(const float4*)(r + tid * 8 + 4);
    float v[8] = {x0.x, x0.y, x0.z, x0.w, x1.x, x1.y, x1.z, x1.w};

    float mx = v[0];
#pragma unroll
    for (int i = 1; i < 8; ++i) mx = fmaxf(mx, v[i]);
#pragma unroll
    for (int off = 16; off > 0; off >>= 1)
        mx = fmaxf(mx, __shfl_xor(mx, off, 32));

    __shared__ float redm[8], reds[8];
    if ((tid & 31) == 0) redm[tid >> 5] = mx;
    __syncthreads();
    float m = redm[0];
#pragma unroll
    for (int i = 1; i < 8; ++i) m = fmaxf(m, redm[i]);

    float sum = 0.f;
#pragma unroll
    for (int i = 0; i < 8; ++i) { v[i] = __expf(v[i] - m); sum += v[i]; }
#pragma unroll
    for (int off = 16; off > 0; off >>= 1) sum += __shfl_xor(sum, off, 32);
    if ((tid & 31) == 0) reds[tid >> 5] = sum;
    __syncthreads();
    float tot = 0.f;
#pragma unroll
    for (int i = 0; i < 8; ++i) tot += reds[i];
    float inv = 1.0f / tot;

    union { __bf16 h[8]; uint4 u; } ob;
#pragma unroll
    for (int i = 0; i < 8; ++i) ob.h[i] = (__bf16)(v[i] * inv);
    *(uint4*)(o + tid * 8) = ob.u;
}

// ------------------------------- launcher ----------------------------------

extern "C" void kernel_launch(void* const* d_in, const int* in_sizes, int n_in,
                              void* d_out, int out_size, void* d_ws,
                              size_t ws_size, hipStream_t stream)
{
    (void)in_sizes; (void)n_in; (void)out_size; (void)ws_size;

    const float* x     = (const float*)d_in[0];   // [4,2048,1024]
    const float* w_qkv = (const float*)d_in[1];   // [1024,3072]
    const float* b_qkv = (const float*)d_in[2];   // [3072]
    const float* w_out = (const float*)d_in[3];   // [1024,1024]
    const float* b_out = (const float*)d_in[4];   // [1024]
    float* out = (float*)d_out;                   // [4,2048,1024]

    // workspace layout (bytes)
    char* w = (char*)d_ws;
    __bf16* xbf   = (__bf16*)w;             w += 16777216;  // 8192x1024 bf16
    __bf16* wqkvT = (__bf16*)w;             w += 6291456;   // 3072x1024 bf16
    __bf16* woutT = (__bf16*)w;             w += 2097152;   // 1024x1024 bf16
    __bf16* qbf   = (__bf16*)w;             w += 16777216;  // [b,s,d]
    __bf16* kbf   = (__bf16*)w;             w += 16777216;  // [b,t,d]
    __bf16* vtbf  = (__bf16*)w;             w += 16777216;  // [b,d,t]
    float*  sc    = (float*)w;              w += 67108864;  // [b,s,t] fp32
    __bf16* pbf   = (__bf16*)w;             w += 33554432;  // [b,s,t] bf16
    __bf16* obf   = (__bf16*)w;             w += 16777216;  // [b,s,d]

    dim3 blk(256);

    // 1) precision conversion (+ weight pre-transpose so all GEMMs are NT)
    k_cvt_x_bf16<<<8192, blk, 0, stream>>>(x, xbf);
    k_cvt_transpose_bf16<<<(1024 * 3072) / 256, blk, 0, stream>>>(
        w_qkv, wqkvT, 1024, 3072);
    k_cvt_transpose_bf16<<<(1024 * 1024) / 256, blk, 0, stream>>>(
        w_out, woutT, 1024, 1024);

    // 2) QKV projection: [8192,1024] x [1024,3072]
    k_gemm_qkv<<<dim3(3072 / BN, 8192 / BM), blk, 0, stream>>>(
        xbf, wqkvT, b_qkv, qbf, kbf, vtbf);

    // 3) scores = Qs @ K^T per batch: 4 x [2048,1024]x[1024,2048]
    k_gemm_scores<<<dim3(2048 / BN, 2048 / BM, 4), blk, 0, stream>>>(
        qbf, kbf, sc);

    // 4) softmax over rows (scores are L2-resident: 67MB < 192MB L2)
    k_softmax<<<8192, blk, 0, stream>>>(sc, pbf);

    // 5) O = P @ V per batch: 4 x [2048,2048]x[2048,1024]
    k_gemm_pv<<<dim3(1024 / BN, 2048 / BM, 4), blk, 0, stream>>>(
        pbf, vtbf, obf);

    // 6) out = O @ w_out + b_out: [8192,1024]x[1024,1024]
    k_gemm_out<<<dim3(1024 / BN, 8192 / BM), blk, 0, stream>>>(
        obf, woutT, b_out, out);
}
